// CausalLocalAttention_71786083386178
// MI455X (gfx1250) — matrix operand order repaired and checked by reading, not compiled, not run on hardware
//
#include <hip/hip_runtime.h>

typedef __attribute__((ext_vector_type(16))) _Float16 v16h;
typedef __attribute__((ext_vector_type(8)))  _Float16 v8h;
typedef __attribute__((ext_vector_type(4)))  _Float16 v4h;
typedef __attribute__((ext_vector_type(8)))  float    v8f;

#define NH   16
#define DH   64
#define WIN  256
#define NWIN 32
#define NKEY 512

// padded LDS row strides (halves) for bank-conflict-free WMMA fragment loads
#define KROW 72
#define VROW 520
#define PROW 72
#define V_OFF (NKEY * KROW)          /* 36864 halves */
#define P_OFF (V_OFF + DH * VROW)    /* 70144 halves */
#define SMEM_HALVES (P_OFF + 16 * 16 * PROW)
#define SMEM_BYTES  (SMEM_HALVES * 2) /* 177152 B */

#define NEGMAX (-3.402823466e38f)
#define ROPE_C 0.28782313662425575f   /* ln(10000)/32 */
#define QSCALE 0.125f                 /* 64^-0.5 */

__device__ __forceinline__ v16h ld_ab(const _Float16* p0, const _Float16* p1) {
  v8h a = *(const v8h*)p0;
  v8h b = *(const v8h*)p1;
  return __builtin_shufflevector(a, b, 0,1,2,3,4,5,6,7,8,9,10,11,12,13,14,15);
}

__global__ __launch_bounds__(512, 1) void local_attn_kernel(
    const float* __restrict__ Q, const float* __restrict__ K,
    const float* __restrict__ V, float* __restrict__ O) {
  const int w = blockIdx.x & (NWIN - 1);
  const int h = blockIdx.x >> 5;
  extern __shared__ _Float16 smem[];
  _Float16* Klds = smem;          // [512][KROW]  rope'd keys, f16, row-major
  _Float16* Vt   = smem + V_OFF;  // [64][VROW]   V transposed, f16
  _Float16* Plds = smem + P_OFF;  // per-wave [16][PROW] P tiles

  const int tid  = threadIdx.x;
  const int lane = tid & 31;
  const int wv   = tid >> 5;      // wave id 0..15 -> query tile
  const int hib  = lane >> 4;     // which 16-lane half
  const int n    = lane & 15;
  const float wf = (float)w;      // RoPE angle depends only on window index

  // ---- Phase A: cooperative load K (rope'd) and V^T into LDS (f32 -> f16) ----
  {
    const int j  = tid;                     // key slot 0..511
    const int tk = w * WIN - WIN + j;       // global key position
    if (tk >= 0) {
      const float* krow = K + ((size_t)tk * NH + h) * DH;
      const float* vrow = V + ((size_t)tk * NH + h) * DH;
#pragma unroll
      for (int d = 0; d < DH; d += 4) {
        float4 kd = *(const float4*)(krow + d);
        float4 vd = *(const float4*)(vrow + d);
        const float f0   = (float)(d >> 1);
        const float inv0 = __expf(-f0 * ROPE_C);
        const float inv1 = __expf(-(f0 + 1.0f) * ROPE_C);
        float s0, c0, s1, c1;
        __sincosf(wf * inv0, &s0, &c0);
        __sincosf(wf * inv1, &s1, &c1);
        v4h kk;
        kk[0] = (_Float16)(kd.x * c0 - kd.y * s0);
        kk[1] = (_Float16)(kd.x * s0 + kd.y * c0);
        kk[2] = (_Float16)(kd.z * c1 - kd.w * s1);
        kk[3] = (_Float16)(kd.z * s1 + kd.w * c1);
        *(v4h*)(Klds + (size_t)j * KROW + d) = kk;
        Vt[(size_t)(d + 0) * VROW + j] = (_Float16)vd.x;
        Vt[(size_t)(d + 1) * VROW + j] = (_Float16)vd.y;
        Vt[(size_t)(d + 2) * VROW + j] = (_Float16)vd.z;
        Vt[(size_t)(d + 3) * VROW + j] = (_Float16)vd.w;
      }
    }
  }

  // ---- Phase B: Q fragments directly in WMMA A-layout (rope + scale) ----
  // A 16x32 f16 layout: lane holds row m=lane%16; elems 0..7 -> K=o..o+7,
  // elems 8..15 -> K=16+o..16+o+7 with o = (lane<16 ? 0 : 8).
  v16h aq0, aq1;
  {
    const int o  = hib ? 8 : 0;
    const int qi = wv * 16 + n;
    const float* qrow = Q + (((size_t)(w * WIN + qi)) * NH + h) * DH;
    _Float16 qh[32];
#pragma unroll
    for (int seg = 0; seg < 4; ++seg) {
      const int d0 = o + seg * 16;
      float x[8];
      *(float4*)&x[0] = *(const float4*)(qrow + d0);
      *(float4*)&x[4] = *(const float4*)(qrow + d0 + 4);
#pragma unroll
      for (int p = 0; p < 4; ++p) {
        const float f   = (float)((d0 >> 1) + p);
        const float inv = __expf(-f * ROPE_C);
        float s, c;
        __sincosf(wf * inv, &s, &c);
        const float xe = x[2 * p], xo = x[2 * p + 1];
        qh[seg * 8 + 2 * p]     = (_Float16)(QSCALE * (xe * c - xo * s));
        qh[seg * 8 + 2 * p + 1] = (_Float16)(QSCALE * (xe * s + xo * c));
      }
    }
#pragma unroll
    for (int e = 0; e < 16; ++e) { aq0[e] = qh[e]; aq1[e] = qh[16 + e]; }
  }

  __syncthreads();

  // ---- streaming flash attention over key chunks of 64 ----
  const v8f z = {0.f, 0.f, 0.f, 0.f, 0.f, 0.f, 0.f, 0.f};
  v8f oacc[4] = {z, z, z, z};
  float mrun[8], lrun[8];
#pragma unroll
  for (int r = 0; r < 8; ++r) { mrun[r] = NEGMAX; lrun[r] = 0.f; }

  const int koff  = hib ? 16 : 0;          // B-fragment K-subblock offset
  const int o2    = hib ? 8 : 0;           // A-fragment K offset
  const int cbeg  = (w == 0) ? 4 : 0;      // skip -1 padded keys of window 0
  const int cend  = (wv * 16 + 271) >> 6;  // last 64-key chunk touching this tile
  const int ibase = wv * 16 + hib * 8 + 256;
  _Float16* pwave = Plds + (size_t)wv * 16 * PROW;

  for (int c = cbeg; c <= cend; ++c) {
    const int jb = c * 64;
    // --- S = Q K^T : 4 n-tiles x (K=64 as 2x32) ---
    v8f s[4] = {z, z, z, z};
#pragma unroll
    for (int nt = 0; nt < 4; ++nt) {
      const _Float16* kr = Klds + (size_t)(jb + nt * 16 + n) * KROW + o2;
      v16h b0 = ld_ab(kr,      kr + 16);
      v16h b1 = ld_ab(kr + 32, kr + 48);
      s[nt] = __builtin_amdgcn_wmma_f32_16x16x32_f16(false, aq0, false, b0, (short)0, s[nt], false, false);
      s[nt] = __builtin_amdgcn_wmma_f32_16x16x32_f16(false, aq1, false, b1, (short)0, s[nt], false, false);
    }
    // --- causal mask + online softmax (row stats per 16-lane half) ---
#pragma unroll
    for (int r = 0; r < 8; ++r) {
      const int irow = ibase + r;                 // query row + 256
#pragma unroll
      for (int nt = 0; nt < 4; ++nt)
        s[nt][r] = (jb + nt * 16 + n > irow) ? NEGMAX : s[nt][r];
      float mx = fmaxf(fmaxf(s[0][r], s[1][r]), fmaxf(s[2][r], s[3][r]));
      mx = fmaxf(mx, __shfl_xor(mx, 1));
      mx = fmaxf(mx, __shfl_xor(mx, 2));
      mx = fmaxf(mx, __shfl_xor(mx, 4));
      mx = fmaxf(mx, __shfl_xor(mx, 8));
      const float mnew = fmaxf(mrun[r], mx);
      const float a = __expf(mrun[r] - mnew);
      mrun[r] = mnew;
      float p[4];
#pragma unroll
      for (int nt = 0; nt < 4; ++nt) p[nt] = __expf(s[nt][r] - mnew);
      float ps = (p[0] + p[1]) + (p[2] + p[3]);
      ps += __shfl_xor(ps, 1);
      ps += __shfl_xor(ps, 2);
      ps += __shfl_xor(ps, 4);
      ps += __shfl_xor(ps, 8);
      lrun[r] = lrun[r] * a + ps;
      oacc[0][r] *= a; oacc[1][r] *= a; oacc[2][r] *= a; oacc[3][r] *= a;
      // P: C-layout -> wave-private LDS tile (row-major 16x64, padded)
      _Float16* pp = pwave + (size_t)(r + hib * 8) * PROW;
      pp[n]      = (_Float16)p[0];
      pp[n + 16] = (_Float16)p[1];
      pp[n + 32] = (_Float16)p[2];
      pp[n + 48] = (_Float16)p[3];
    }
    // --- O += P V : read P back in A-layout, V^T as B fragments ---
    {
      const _Float16* pr = pwave + (size_t)n * PROW;
      v16h ap0 = ld_ab(pr + koff,      pr + 8 + koff);
      v16h ap1 = ld_ab(pr + 32 + koff, pr + 40 + koff);
#pragma unroll
      for (int dt = 0; dt < 4; ++dt) {
        const _Float16* vr = Vt + (size_t)(dt * 16 + n) * VROW + jb + koff;
        v16h bv0 = ld_ab(vr,      vr + 8);
        v16h bv1 = ld_ab(vr + 32, vr + 40);
        oacc[dt] = __builtin_amdgcn_wmma_f32_16x16x32_f16(false, ap0, false, bv0, (short)0, oacc[dt], false, false);
        oacc[dt] = __builtin_amdgcn_wmma_f32_16x16x32_f16(false, ap1, false, bv1, (short)0, oacc[dt], false, false);
      }
    }
  }

  // ---- normalize + store fp32 output (B,T,H,D) ----
#pragma unroll
  for (int r = 0; r < 8; ++r) {
    const float inv = 1.0f / lrun[r];
    const size_t t = (size_t)w * WIN + wv * 16 + r + hib * 8;
    float* op = O + (t * NH + h) * DH + n;
    op[0]  = oacc[0][r] * inv;
    op[16] = oacc[1][r] * inv;
    op[32] = oacc[2][r] * inv;
    op[48] = oacc[3][r] * inv;
  }
}

extern "C" void kernel_launch(void* const* d_in, const int* in_sizes, int n_in,
                              void* d_out, int out_size, void* d_ws, size_t ws_size,
                              hipStream_t stream) {
  (void)in_sizes; (void)n_in; (void)d_ws; (void)ws_size; (void)out_size;
  const float* q = (const float*)d_in[0];
  const float* k = (const float*)d_in[1];
  const float* v = (const float*)d_in[2];
  float* out = (float*)d_out;
  dim3 grid(NH * NWIN), block(512);
  local_attn_kernel<<<grid, block, SMEM_BYTES, stream>>>(q, k, v, out);
}
